// VariationalLinear_56573309224151
// MI455X (gfx1250) — matrix operand order repaired and checked by reading, not compiled, not run on hardware
//
#include <hip/hip_runtime.h>
#include <hip/hip_bf16.h>

#define IN_F  4096
#define OUT_F 4096
#define BATCH 1024

typedef __attribute__((ext_vector_type(16))) __bf16 bf16x16;
typedef __attribute__((ext_vector_type(8)))  __bf16 bf16x8;
typedef __attribute__((ext_vector_type(4)))  __bf16 bf16x4;
typedef __attribute__((ext_vector_type(8)))  float  v8f;
typedef __attribute__((ext_vector_type(4)))  float  v4f;

union Frag {
    bf16x16 v;
    struct { bf16x8 lo; bf16x8 hi; } p;
};

__device__ __forceinline__ void split1(float w, __bf16& h, __bf16& l) {
    h = (__bf16)w;
    l = (__bf16)(w - (float)h);
}

// ---------- Phase 1a: sample weight W = mu + eps*sigma, split to bf16 hi/lo ----------
__global__ __launch_bounds__(256) void vl_split_w(
    const float* __restrict__ mu, const float* __restrict__ sg,
    const float* __restrict__ ep,
    __bf16* __restrict__ hi, __bf16* __restrict__ lo, long n)
{
    long i = ((long)blockIdx.x * 256 + threadIdx.x) * 4;
    if (i >= n) return;
    v4f m = *(const v4f*)(mu + i);
    v4f s = *(const v4f*)(sg + i);
    v4f e = *(const v4f*)(ep + i);
    bf16x4 h, l;
#pragma unroll
    for (int j = 0; j < 4; ++j) {
        float w = m[j] + e[j] * s[j];
        __bf16 hh, ll; split1(w, hh, ll);
        h[j] = hh; l[j] = ll;
    }
    *(bf16x4*)(hi + i) = h;
    *(bf16x4*)(lo + i) = l;
}

// ---------- Phase 1b: split x to bf16 hi/lo ----------
__global__ __launch_bounds__(256) void vl_split_x(
    const float* __restrict__ x,
    __bf16* __restrict__ hi, __bf16* __restrict__ lo, long n)
{
    long i = ((long)blockIdx.x * 256 + threadIdx.x) * 4;
    if (i >= n) return;
    v4f xv = *(const v4f*)(x + i);
    bf16x4 h, l;
#pragma unroll
    for (int j = 0; j < 4; ++j) {
        __bf16 hh, ll; split1(xv[j], hh, ll);
        h[j] = hh; l[j] = ll;
    }
    *(bf16x4*)(hi + i) = h;
    *(bf16x4*)(lo + i) = l;
}

// ---------- Phase 1c: combined bias = bias_mu + eps_b * bias_sigma ----------
__global__ __launch_bounds__(256) void vl_bias(
    const float* __restrict__ bmu, const float* __restrict__ bsg,
    const float* __restrict__ bep, float* __restrict__ bc, int n)
{
    int i = blockIdx.x * 256 + threadIdx.x;
    if (i < n) bc[i] = bmu[i] + bep[i] * bsg[i];
}

// ---------- Phase 2: out = x @ W^T + bias via bf16x3 WMMA ----------
// Block: 256 threads = 8 waves (4 M-waves x 2 N-waves). WG tile 128(M) x 64(N).
// Wave tile: 32x32 = 2x2 WMMA 16x16 tiles. K stepped by 32.
__global__ __launch_bounds__(256) void vl_gemm(
    const __bf16* __restrict__ Xhi, const __bf16* __restrict__ Xlo,
    const __bf16* __restrict__ Whi, const __bf16* __restrict__ Wlo,
    const float*  __restrict__ biasC, float* __restrict__ out)
{
    const int lane = threadIdx.x & 31;
    const int wave = threadIdx.x >> 5;
    const int wm   = wave >> 1;          // 0..3
    const int wn   = wave & 1;           // 0..1
    const int r    = lane & 15;
    const int half = lane >> 4;          // 0 or 1

    const int m0 = blockIdx.y * 128 + wm * 32;
    const int n0 = blockIdx.x * 64  + wn * 32;

    // A fragment (16x32, 16-bit): lane<16 holds K 0..15 contiguous, lane>=16 K 16..31
    // B fragment (32x16, 16-bit): lane<16 holds K 0..15 contiguous, lane>=16 K 16..31; col = lane&15
    const __bf16* xh[2]; const __bf16* xl[2];
    const __bf16* wh[2]; const __bf16* wl[2];
#pragma unroll
    for (int t = 0; t < 2; ++t) {
        long arow = (long)(m0 + 16 * t + r) * IN_F + half * 16;
        xh[t] = Xhi + arow;
        xl[t] = Xlo + arow;
        long brow = (long)(n0 + 16 * t + r) * IN_F + half * 16;
        wh[t] = Whi + brow;
        wl[t] = Wlo + brow;
    }

    v8f acc[2][2] = {};

    for (int kk = 0; kk < IN_F; kk += 32) {
        Frag ah[2], al[2], bh[2], bl[2];
#pragma unroll
        for (int t = 0; t < 2; ++t) {
            ah[t].p.lo = *(const bf16x8*)(xh[t] + kk);
            ah[t].p.hi = *(const bf16x8*)(xh[t] + kk + 8);
            al[t].p.lo = *(const bf16x8*)(xl[t] + kk);
            al[t].p.hi = *(const bf16x8*)(xl[t] + kk + 8);
            bh[t].v    = *(const bf16x16*)(wh[t] + kk);
            bl[t].v    = *(const bf16x16*)(wl[t] + kk);
        }
        // hint the K-step four ahead (kk+128) of the streamed B operand into the cache hierarchy
        __builtin_prefetch(wh[0] + kk + 128, 0, 0);
        __builtin_prefetch(wl[0] + kk + 128, 0, 0);

#pragma unroll
        for (int mi = 0; mi < 2; ++mi)
#pragma unroll
            for (int ni = 0; ni < 2; ++ni) {
                acc[mi][ni] = __builtin_amdgcn_wmma_f32_16x16x32_bf16(
                    false, ah[mi].v, false, bh[ni].v, (short)0, acc[mi][ni], false, false);
                acc[mi][ni] = __builtin_amdgcn_wmma_f32_16x16x32_bf16(
                    false, ah[mi].v, false, bl[ni].v, (short)0, acc[mi][ni], false, false);
                acc[mi][ni] = __builtin_amdgcn_wmma_f32_16x16x32_bf16(
                    false, al[mi].v, false, bh[ni].v, (short)0, acc[mi][ni], false, false);
            }
    }

    // C/D layout: vector elem j -> row (half*8 + j), col = lane&15 (+tile offsets)
#pragma unroll
    for (int mi = 0; mi < 2; ++mi)
#pragma unroll
        for (int ni = 0; ni < 2; ++ni) {
            int col = n0 + 16 * ni + r;
            float bz = biasC[col];
            float* o = out + (long)(m0 + 16 * mi + half * 8) * OUT_F + col;
#pragma unroll
            for (int j = 0; j < 8; ++j)
                o[(long)j * OUT_F] = acc[mi][ni][j] + bz;
        }
}

// ---------- Fallback (only if workspace is too small): naive fp32 ----------
__global__ __launch_bounds__(256) void vl_naive(
    const float* __restrict__ x,
    const float* __restrict__ wmu, const float* __restrict__ wsg,
    const float* __restrict__ bmu, const float* __restrict__ bsg,
    const float* __restrict__ epw, const float* __restrict__ epb,
    float* __restrict__ out)
{
    int o = blockIdx.x * 256 + threadIdx.x;
    int b = blockIdx.y;
    if (o >= OUT_F) return;
    const float* xr = x   + (long)b * IN_F;
    const float* mr = wmu + (long)o * IN_F;
    const float* sr = wsg + (long)o * IN_F;
    const float* er = epw + (long)o * IN_F;
    float acc = 0.f;
    for (int k = 0; k < IN_F; ++k)
        acc += xr[k] * (mr[k] + er[k] * sr[k]);
    out[(long)b * OUT_F + o] = acc + bmu[o] + epb[o] * bsg[o];
}

extern "C" void kernel_launch(void* const* d_in, const int* in_sizes, int n_in,
                              void* d_out, int out_size, void* d_ws, size_t ws_size,
                              hipStream_t stream)
{
    (void)in_sizes; (void)n_in; (void)out_size;
    const float* x   = (const float*)d_in[0];
    const float* wmu = (const float*)d_in[1];
    const float* wsg = (const float*)d_in[2];
    const float* bmu = (const float*)d_in[3];
    const float* bsg = (const float*)d_in[4];
    const float* epw = (const float*)d_in[5];
    const float* epb = (const float*)d_in[6];
    float* out = (float*)d_out;

    const long WN = (long)OUT_F * IN_F;       // 16,777,216
    const long XN = (long)BATCH * IN_F;       //  4,194,304
    const size_t need = (size_t)(2 * WN + 2 * XN) * sizeof(__bf16) + OUT_F * sizeof(float);

    if (ws_size < need) {
        dim3 g(OUT_F / 256, BATCH);
        vl_naive<<<g, 256, 0, stream>>>(x, wmu, wsg, bmu, bsg, epw, epb, out);
        return;
    }

    char* ws = (char*)d_ws;
    __bf16* Whi = (__bf16*)(ws);
    __bf16* Wlo = (__bf16*)(ws + (size_t)WN * 2);
    __bf16* Xhi = (__bf16*)(ws + (size_t)WN * 4);
    __bf16* Xlo = (__bf16*)(ws + (size_t)WN * 4 + (size_t)XN * 2);
    float*  bc  = (float*) (ws + (size_t)WN * 4 + (size_t)XN * 4);

    vl_split_w<<<(int)(WN / 1024), 256, 0, stream>>>(wmu, wsg, epw, Whi, Wlo, WN);
    vl_split_x<<<(int)(XN / 1024), 256, 0, stream>>>(x, Xhi, Xlo, XN);
    vl_bias   <<<OUT_F / 256, 256, 0, stream>>>(bmu, bsg, epb, bc, OUT_F);

    dim3 grid(OUT_F / 64, BATCH / 128);   // (64, 8)
    vl_gemm<<<grid, 256, 0, stream>>>(Xhi, Xlo, Whi, Wlo, bc, out);
}
